// SimpleSDPAAttention_20598663152308
// MI455X (gfx1250) — matrix operand order repaired and checked by reading, not compiled, not run on hardware
//
#include <hip/hip_runtime.h>

#define S_LEN   2048
#define NH      32
#define NKV     8
#define DH      128
#define SCALE_F 0.08838834764831845f

#define BLOCK_M 128   // queries per workgroup (8 waves x 16 rows)
#define BLOCK_N 32    // keys per inner tile

typedef __attribute__((ext_vector_type(16))) _Float16 v16h;
typedef __attribute__((ext_vector_type(8)))  _Float16 v8h;
typedef __attribute__((ext_vector_type(8)))  float    v8f;
typedef __attribute__((ext_vector_type(4)))  float    v4f;

// ds_swizzle_b32 bitmask mode: and_mask=[4:0]=0x1f, or_mask=[9:5]=0, xor_mask=[14:10].
// xor masks 1/2/4/8 stay within each 16-lane half -> per-row butterfly reduction
// with an immediate-encoded swizzle (no address VALU, unlike ds_bpermute).
#define SWZ_XOR(x, m) \
    __int_as_float(__builtin_amdgcn_ds_swizzle(__float_as_int(x), (((m) << 10) | 0x1f)))

__device__ __forceinline__ float rowmax16(float x) {
    x = fmaxf(x, SWZ_XOR(x, 1));
    x = fmaxf(x, SWZ_XOR(x, 2));
    x = fmaxf(x, SWZ_XOR(x, 4));
    x = fmaxf(x, SWZ_XOR(x, 8));
    return x;
}
__device__ __forceinline__ float rowsum16(float x) {
    x += SWZ_XOR(x, 1);
    x += SWZ_XOR(x, 2);
    x += SWZ_XOR(x, 4);
    x += SWZ_XOR(x, 8);
    return x;
}

__global__ __launch_bounds__(256, 1)
void SimpleSDPAAttention_fa_fwd(const float* __restrict__ q_g,
                                const float* __restrict__ k_g,
                                const float* __restrict__ v_g,
                                float* __restrict__ o_g)
{
    // K tile row-major [key][d], V tile transposed [d][key], per-wave P scratch [row][col]
    __shared__ _Float16 Ks[BLOCK_N * DH];
    __shared__ _Float16 Vt[DH * BLOCK_N];
    __shared__ _Float16 Ps[8][16 * BLOCK_N];

    const int tid  = threadIdx.x;
    const int lane = tid & 31;
    const int wave = tid >> 5;
    const int ln16 = lane & 15;
    const int half = lane >> 4;

    const int h     = blockIdx.y;        // query head
    const int kvh   = h >> 2;            // GQA: 4 query heads per kv head
    const int m0    = blockIdx.x * BLOCK_M;
    const int qbase = m0 + wave * 16;
    const int qrow  = qbase + ln16;      // A-fragment row for this lane

    // ---- Q: 4 A-fragments (16x32 f16 each); fold softmax SCALE into Q here ----
    v16h qf[4];
    {
        const float* qp = q_g + (size_t)qrow * (NH * DH) + (size_t)h * DH;
        #pragma unroll
        for (int c = 0; c < 4; ++c) {
            const int dbase = c * 32 + 16 * half;          // A order used here (B matches): K = 16*half + 0..15
            v4f x0 = *(const v4f*)(qp + dbase);
            v4f x1 = *(const v4f*)(qp + dbase + 4);
            v4f y0 = *(const v4f*)(qp + dbase + 8);
            v4f y1 = *(const v4f*)(qp + dbase + 12);
            v16h f;
            #pragma unroll
            for (int i = 0; i < 4; ++i) {
                f[i]      = (_Float16)(x0[i] * SCALE_F);
                f[4 + i]  = (_Float16)(x1[i] * SCALE_F);
                f[8 + i]  = (_Float16)(y0[i] * SCALE_F);
                f[12 + i] = (_Float16)(y1[i] * SCALE_F);
            }
            qf[c] = f;
        }
    }

    // ---- accumulator state ----
    v8f Oa[8];
    #pragma unroll
    for (int f = 0; f < 8; ++f) Oa[f] = (v8f){};
    float m_i[8], l_i[8];
    #pragma unroll
    for (int r = 0; r < 8; ++r) { m_i[r] = -__builtin_inff(); l_i[r] = 0.0f; }

    // cooperative loader mapping: 8 threads per key row, 16 contiguous d each
    const int krow = tid >> 3;            // 0..31
    const int d0   = (tid & 7) * 16;      // 0,16,...,112

    const int jend = m0 + BLOCK_M;        // causal bound for this query tile
    for (int j0 = 0; j0 < jend; j0 += BLOCK_N) {
        __syncthreads();   // previous iteration done reading LDS tiles

        // ---- stage K (row-major) and V (transposed) tiles as f16 in LDS ----
        {
            const float* kp = k_g + (size_t)(j0 + krow) * (NKV * DH) + (size_t)kvh * DH + d0;
            v4f a0 = *(const v4f*)(kp + 0);
            v4f a1 = *(const v4f*)(kp + 4);
            v4f a2 = *(const v4f*)(kp + 8);
            v4f a3 = *(const v4f*)(kp + 12);
            v16h kv;
            #pragma unroll
            for (int i = 0; i < 4; ++i) {
                kv[i]      = (_Float16)a0[i];
                kv[4 + i]  = (_Float16)a1[i];
                kv[8 + i]  = (_Float16)a2[i];
                kv[12 + i] = (_Float16)a3[i];
            }
            *(v16h*)&Ks[krow * DH + d0] = kv;

            const float* vp = v_g + (size_t)(j0 + krow) * (NKV * DH) + (size_t)kvh * DH + d0;
            v4f b0 = *(const v4f*)(vp + 0);
            v4f b1 = *(const v4f*)(vp + 4);
            v4f b2 = *(const v4f*)(vp + 8);
            v4f b3 = *(const v4f*)(vp + 12);
            #pragma unroll
            for (int i = 0; i < 4; ++i) {
                Vt[(d0 + i)      * BLOCK_N + krow] = (_Float16)b0[i];
                Vt[(d0 + 4 + i)  * BLOCK_N + krow] = (_Float16)b1[i];
                Vt[(d0 + 8 + i)  * BLOCK_N + krow] = (_Float16)b2[i];
                Vt[(d0 + 12 + i) * BLOCK_N + krow] = (_Float16)b3[i];
            }
            if (j0 + BLOCK_N < jend) {   // global_prefetch_b8 for next tile
                __builtin_prefetch(kp + (size_t)BLOCK_N * NKV * DH, 0, 3);
                __builtin_prefetch(vp + (size_t)BLOCK_N * NKV * DH, 0, 3);
            }
        }
        __syncthreads();

        // Wave-uniform causal classification of this tile vs this wave's 16 rows.
        if (j0 > qbase + 15) continue;                 // fully masked: skip compute
        const bool needMask = (j0 + BLOCK_N - 1) > qbase;  // diagonal band only

        // ---- preload all K B-fragments, then run the QK^T WMMA chain ----
        v16h kf[8];
        #pragma unroll
        for (int c = 0; c < 4; ++c) {
            kf[2 * c]     = *(const v16h*)&Ks[(ln16)      * DH + c * 32 + 16 * half];
            kf[2 * c + 1] = *(const v16h*)&Ks[(16 + ln16) * DH + c * 32 + 16 * half];
        }
        v8f s0 = (v8f){};
        v8f s1 = (v8f){};
        #pragma unroll
        for (int c = 0; c < 4; ++c) {
            s0 = __builtin_amdgcn_wmma_f32_16x16x32_f16(false, qf[c], false, kf[2 * c],     (short)0, s0, false, false);
            s1 = __builtin_amdgcn_wmma_f32_16x16x32_f16(false, qf[c], false, kf[2 * c + 1], (short)0, s1, false, false);
        }

        // ---- issue V B-fragment loads now; they complete under the softmax VALU ----
        v16h vf[8];
        #pragma unroll
        for (int f = 0; f < 8; ++f)
            vf[f] = *(const v16h*)&Vt[(f * 16 + ln16) * BLOCK_N + 16 * half];

        // ---- online softmax over this 16x32 tile; emit P (f16) to LDS ----
        #pragma unroll
        for (int r = 0; r < 8; ++r) {
            const int row = r + 8 * half;       // C layout row owned by this lane
            float a = s0[r];
            float b = s1[r];
            if (needMask) {
                const int qi = qbase + row;
                const int c0 = j0 + ln16;
                a = (c0      <= qi) ? a : -__builtin_inff();
                b = (c0 + 16 <= qi) ? b : -__builtin_inff();
            }
            const float mx   = rowmax16(fmaxf(a, b));
            const float mnew = fmaxf(m_i[r], mx);
            const float pa   = __expf(a - mnew);
            const float pb   = __expf(b - mnew);
            const float rs   = rowsum16(pa + pb);
            const float alpha = __expf(m_i[r] - mnew);
            m_i[r] = mnew;
            l_i[r] = l_i[r] * alpha + rs;
            #pragma unroll
            for (int f = 0; f < 8; ++f) Oa[f][r] *= alpha;
            Ps[wave][row * BLOCK_N + ln16]      = (_Float16)pa;
            Ps[wave][row * BLOCK_N + 16 + ln16] = (_Float16)pb;
        }

        // wave-internal C->A transpose through LDS: drain DS before re-reading
        asm volatile("s_wait_dscnt 0" ::: "memory");

        // P as A-fragment (16x32): row = lane%16, K = 16*half+0..15
        v8h plo = *(const v8h*)&Ps[wave][ln16 * BLOCK_N + 16 * half];
        v8h phi = *(const v8h*)&Ps[wave][ln16 * BLOCK_N + 16 * half + 8];
        v16h pf = __builtin_shufflevector(plo, phi, 0,1,2,3,4,5,6,7,8,9,10,11,12,13,14,15);

        // ---- O += P V : 8 output d-subtiles, operands already resident ----
        #pragma unroll
        for (int f = 0; f < 8; ++f)
            Oa[f] = __builtin_amdgcn_wmma_f32_16x16x32_f16(false, pf, false, vf[f], (short)0, Oa[f], false, false);
    }

    // ---- epilogue: normalize and store fp32 ----
    #pragma unroll
    for (int r = 0; r < 8; ++r) {
        const float inv = 1.0f / l_i[r];
        const int row = r + 8 * half;
        const int qi  = qbase + row;
        float* op = o_g + (size_t)qi * (NH * DH) + (size_t)h * DH + ln16;
        #pragma unroll
        for (int f = 0; f < 8; ++f)
            op[f * 16] = Oa[f][r] * inv;
    }
}

extern "C" void kernel_launch(void* const* d_in, const int* in_sizes, int n_in,
                              void* d_out, int out_size, void* d_ws, size_t ws_size,
                              hipStream_t stream) {
    (void)in_sizes; (void)n_in; (void)out_size; (void)d_ws; (void)ws_size;
    const float* q = (const float*)d_in[0];
    const float* k = (const float*)d_in[1];
    const float* v = (const float*)d_in[2];
    float* out = (float*)d_out;
    dim3 grid(S_LEN / BLOCK_M, NH);
    SimpleSDPAAttention_fa_fwd<<<grid, dim3(256), 0, stream>>>(q, k, v, out);
}
